// MultiHeadSelfAttention_76785425318009
// MI455X (gfx1250) — matrix operand order repaired and checked by reading, not compiled, not run on hardware
//
#include <hip/hip_runtime.h>
#include <hip/hip_bf16.h>

// ---------------------------------------------------------------------------
// MI455X (gfx1250) multi-head self-attention, bf16 WMMA (f32 accumulate).
// B=4, S=2048, D=1024, H=16, dk=64.
// ---------------------------------------------------------------------------

#define D_MODEL 1024
#define NHEADS  16
#define SEQ     2048
#define DK      64
#define BATCH   4

#define AS1 __attribute__((address_space(1)))
#define AS3 __attribute__((address_space(3)))

#ifdef __has_builtin
#  if __has_builtin(__builtin_amdgcn_global_load_async_to_lds_b128)
#    define HAVE_ASYNC_LDS 1
#  endif
#endif
#ifndef HAVE_ASYNC_LDS
#  define HAVE_ASYNC_LDS 0
#endif

typedef __attribute__((ext_vector_type(16))) __bf16 bf16x16;
typedef __attribute__((ext_vector_type(8)))  float  f32x8;
typedef __attribute__((ext_vector_type(4)))  int    i32x4;

union BFrag { bf16x16 v; unsigned short u[16]; };

// fp32 -> bf16 round-to-nearest-even (bit pattern)
__device__ __forceinline__ unsigned short f2bf(float f) {
    union { float f; unsigned u; } a; a.f = f;
    unsigned u = a.u;
    unsigned r = u + 0x7FFFu + ((u >> 16) & 1u);
    return (unsigned short)(r >> 16);
}

__device__ __forceinline__ f32x8 wmma_bf16(const BFrag& a, const BFrag& b, f32x8 c) {
    // D = A(16x32 bf16) x B(32x16 bf16) + C(16x16 f32)
    return __builtin_amdgcn_wmma_f32_16x16x32_bf16(
        /*neg_a=*/false, a.v, /*neg_b=*/false, b.v,
        /*c_mod=*/(short)0, c, /*reuse_a=*/false, /*reuse_b=*/false);
}

#if HAVE_ASYNC_LDS
__device__ __forceinline__ void async_ld_b128(const void* g, void* l) {
    // per-lane: LDS[l] = MEM[g], 16 bytes, tracked by ASYNCcnt
    __builtin_amdgcn_global_load_async_to_lds_b128((AS1 i32x4*)(void*)g,
                                                   (AS3 i32x4*)(void*)l, 0, 0);
}
__device__ __forceinline__ void async_wait0() {
#  if defined(__has_builtin) && __has_builtin(__builtin_amdgcn_s_wait_asynccnt)
    __builtin_amdgcn_s_wait_asynccnt(0);
#  else
    asm volatile("s_wait_asynccnt 0" ::: "memory");
#  endif
}
#endif

// 16-lane-group reductions (xor masks 1..8 never cross the 16-lane half)
__device__ __forceinline__ float rmax16(float x) {
    #pragma unroll
    for (int m = 8; m >= 1; m >>= 1) x = fmaxf(x, __shfl_xor(x, m, 32));
    return x;
}
__device__ __forceinline__ float rsum16(float x) {
    #pragma unroll
    for (int m = 8; m >= 1; m >>= 1) x += __shfl_xor(x, m, 32);
    return x;
}

// ---------------------------------------------------------------------------
// Tiled GEMM: out[row][col] = sum_i W[col][i] * in[row][i]
// MODE 0: in = fp32 (x), out = bf16 in head layout [B,H,S,64], scaled
// MODE 1: in = bf16 (ctx), out = fp32 flat [B,S,D]
// Block: 256 threads = 8 waves, tile 64(M) x 128(N), K-step 32.
// ---------------------------------------------------------------------------
template <int MODE>
__global__ __launch_bounds__(256) void gemm_kernel(const void* __restrict__ inp,
                                                   const float* __restrict__ W,
                                                   void* __restrict__ outp,
                                                   float scale) {
    __shared__ unsigned short As[64][40];    // 64 rows x 32 k (bf16), +8 pad
    __shared__ unsigned short Bs[128][40];   // 128 n  x 32 k (bf16), +8 pad

    const int tid  = threadIdx.x;
    const int lane = tid & 31;
    const int wave = tid >> 5;
    const int mw = wave >> 1, nw = wave & 1;
    const int nb = blockIdx.x & 7;            // 8 col-blocks of 128
    const int mb = blockIdx.x >> 3;           // 128 row-blocks of 64
    const int row0 = mb * 64, col0 = nb * 128;
    const int lm = lane & 15, hs = lane >> 4;

    f32x8 acc[4] = {};
    bool used_async = false;

    for (int k0 = 0; k0 < D_MODEL; k0 += 32) {
        // ---- stage A tile (64x32) as bf16 ----
        {
            const int r = tid >> 2, c = (tid & 3) * 8;
            if (MODE == 0) {
                const float* X = (const float*)inp + (size_t)(row0 + r) * D_MODEL + k0 + c;
                #pragma unroll
                for (int j = 0; j < 8; ++j) As[r][c + j] = f2bf(X[j]);
            } else {
                const unsigned short* X =
                    (const unsigned short*)inp + (size_t)(row0 + r) * D_MODEL + k0 + c;
#if HAVE_ASYNC_LDS
                async_ld_b128(X, &As[r][c]);          // 8 bf16 = 16B per lane
                used_async = true;
#else
                #pragma unroll
                for (int j = 0; j < 8; ++j) As[r][c + j] = X[j];
#endif
            }
        }
        // ---- stage B tile: Bs[n][i] = bf16(W[col0+n][k0+i]) ----
        {
            const int n = tid >> 1, io = (tid & 1) * 16;
            const float* Wp = W + (size_t)(col0 + n) * D_MODEL + k0 + io;
            #pragma unroll
            for (int j = 0; j < 16; ++j) Bs[n][io + j] = f2bf(Wp[j]);
        }
#if HAVE_ASYNC_LDS
        if (used_async) async_wait0();
#endif
        (void)used_async;
        __syncthreads();

        // ---- A fragment (16x32): lane M = mw*16+lm; K contiguous (k = hs*16 + e), as in B ----
        BFrag a;
        {
            const int M = mw * 16 + lm;
            #pragma unroll
            for (int j = 0; j < 8; ++j) {
                const int kb = hs * 16 + 2 * j;
                a.u[2 * j]     = As[M][kb];
                a.u[2 * j + 1] = As[M][kb + 1];
            }
        }
        // ---- 4 B fragments + WMMA ----
        #pragma unroll
        for (int t = 0; t < 4; ++t) {
            BFrag b;
            const int N = nw * 64 + t * 16 + lm;
            #pragma unroll
            for (int e = 0; e < 16; ++e) b.u[e] = Bs[N][hs * 16 + e];
            acc[t] = wmma_bf16(a, b, acc[t]);
        }
        __syncthreads();
    }

    // ---- epilogue (C layout: VGPR r -> row M = r + 8*hs, col = lm) ----
    #pragma unroll
    for (int t = 0; t < 4; ++t) {
        #pragma unroll
        for (int r = 0; r < 8; ++r) {
            const int grow = row0 + mw * 16 + r + 8 * hs;   // global row over B*S
            const int gcol = col0 + nw * 64 + t * 16 + lm;  // output channel
            const float val = acc[t][r] * scale;
            if (MODE == 0) {
                const int b = grow >> 11, s = grow & (SEQ - 1);
                const int h = gcol >> 6,  dc = gcol & 63;
                ((unsigned short*)outp)[((size_t)(b * NHEADS + h) * SEQ + s) * DK + dc] =
                    f2bf(val);
            } else {
                ((float*)outp)[(size_t)grow * D_MODEL + gcol] = val;
            }
        }
    }
}

// ---------------------------------------------------------------------------
// Flash-attention (causal, online softmax in log2 domain), bf16 WMMA.
// Block: 128 threads = 4 waves; each wave owns 16 query rows.
// Block streams 32-key K/V tiles through LDS via async-to-LDS b128.
// ---------------------------------------------------------------------------
__global__ __launch_bounds__(128) void attn_kernel(const unsigned short* __restrict__ q,
                                                   const unsigned short* __restrict__ k,
                                                   const unsigned short* __restrict__ v,
                                                   unsigned short* __restrict__ ctx) {
    __shared__ unsigned short Ks[32][72];      // [key][dk] row-major (+pad, 16B-aligned rows)
    __shared__ unsigned short Vs[32][72];      // [key][dk] row-major
    __shared__ unsigned short Ps[4][16][40];   // per-wave P re-layout scratch

    const int tid  = threadIdx.x;
    const int lane = tid & 31;
    const int wave = tid >> 5;
    const int lm = lane & 15, hs = lane >> 4;

    const int QB = blockIdx.x & (SEQ / 64 - 1);
    const int hh = (blockIdx.x / (SEQ / 64)) & (NHEADS - 1);
    const int bb = blockIdx.x / ((SEQ / 64) * NHEADS);
    const size_t head_off = (size_t)(bb * NHEADS + hh) * SEQ * DK;
    const int q0b = QB * 64;
    const int q0w = q0b + wave * 16;

    // Q fragments: dk=64 -> two 16x32 A fragments (1/sqrt(dk)*log2e folded in)
    BFrag aq[2];
    {
        const unsigned short* qp = q + head_off + (size_t)(q0w + lm) * DK;
        #pragma unroll
        for (int kk = 0; kk < 2; ++kk)
            #pragma unroll
            for (int j = 0; j < 8; ++j) {
                const int kb = kk * 32 + hs * 16 + 2 * j;
                aq[kk].u[2 * j]     = qp[kb];
                aq[kk].u[2 * j + 1] = qp[kb + 1];
            }
    }

    f32x8 o[4] = {};
    float m[8], l[8];
    #pragma unroll
    for (int r = 0; r < 8; ++r) { m[r] = -__builtin_inff(); l[r] = 0.f; }

    const int nkt = q0b / 32 + 2;   // causal bound for this block
    for (int kt = 0; kt < nkt; ++kt) {
        const int k0 = kt * 32;
        // ---- cooperative staging of K and V tiles (row-major) ----
        {
            const int key = tid >> 2, doff = (tid & 3) * 16;
            const unsigned short* kp = k + head_off + (size_t)(k0 + key) * DK + doff;
            const unsigned short* vp = v + head_off + (size_t)(k0 + key) * DK + doff;
#if HAVE_ASYNC_LDS
            async_ld_b128(kp,     &Ks[key][doff]);
            async_ld_b128(kp + 8, &Ks[key][doff + 8]);
            async_ld_b128(vp,     &Vs[key][doff]);
            async_ld_b128(vp + 8, &Vs[key][doff + 8]);
#else
            #pragma unroll
            for (int j = 0; j < 16; ++j) Ks[key][doff + j] = kp[j];
            #pragma unroll
            for (int j = 0; j < 16; ++j) Vs[key][doff + j] = vp[j];
#endif
            if (kt + 1 < nkt) {                        // prefetch next tile
                __builtin_prefetch(kp + 32 * DK, 0, 0);
                __builtin_prefetch(vp + 32 * DK, 0, 0);
            }
        }
#if HAVE_ASYNC_LDS
        async_wait0();
#endif
        __syncthreads();

        if (k0 <= q0w + 15) {
            // ---- scores: two 16x16 tiles over the 32 keys ----
            f32x8 sc[2];
            #pragma unroll
            for (int ns = 0; ns < 2; ++ns) {
                f32x8 c = {};
                #pragma unroll
                for (int kk = 0; kk < 2; ++kk) {
                    BFrag bk;                          // B[k][n] = K[key=n][dk=k]
                    const int key = ns * 16 + lm;
                    #pragma unroll
                    for (int e = 0; e < 16; ++e)
                        bk.u[e] = Ks[key][kk * 32 + hs * 16 + e];
                    c = wmma_bf16(aq[kk], bk, c);
                }
                sc[ns] = c;
            }

            // ---- causal mask + online softmax (log2 domain; row M = r+8*hs) ----
            float alpha[8];
            #pragma unroll
            for (int r = 0; r < 8; ++r) {
                const int   qrow = q0w + r + 8 * hs;
                const float NEG  = -__builtin_inff();
                float s0 = ((k0 + lm)      <= qrow) ? sc[0][r] : NEG;
                float s1 = ((k0 + 16 + lm) <= qrow) ? sc[1][r] : NEG;
                float mx = rmax16(fmaxf(s0, s1));
                float mn = fmaxf(m[r], mx);
                alpha[r] = __builtin_amdgcn_exp2f(m[r] - mn);
                float p0 = __builtin_amdgcn_exp2f(s0 - mn);
                float p1 = __builtin_amdgcn_exp2f(s1 - mn);
                float rs = rsum16(p0 + p1);
                l[r] = l[r] * alpha[r] + rs;
                m[r] = mn;
                // park P in LDS (bf16) for C->A fragment re-layout
                Ps[wave][r + 8 * hs][lm]      = f2bf(p0);
                Ps[wave][r + 8 * hs][16 + lm] = f2bf(p1);
            }
            // rescale running output accumulators
            #pragma unroll
            for (int t = 0; t < 4; ++t)
                #pragma unroll
                for (int r = 0; r < 8; ++r) o[t][r] *= alpha[r];

            // ---- P (16x32, A layout) x V (32x64) ----
            BFrag ap;
            #pragma unroll
            for (int j = 0; j < 8; ++j) {
                const int kb = hs * 16 + 2 * j;
                ap.u[2 * j]     = Ps[wave][lm][kb];
                ap.u[2 * j + 1] = Ps[wave][lm][kb + 1];
            }
            #pragma unroll
            for (int t = 0; t < 4; ++t) {
                BFrag bv;                              // B[k][n] = V[key=k][dk=n]
                #pragma unroll
                for (int e = 0; e < 16; ++e)
                    bv.u[e] = Vs[hs * 16 + e][t * 16 + lm];
                o[t] = wmma_bf16(ap, bv, o[t]);
            }
        }
        __syncthreads();
    }

    // ---- normalize + write ctx (bf16, [B,S,D] with D = h*64 + dk) ----
    float inv[8];
    #pragma unroll
    for (int r = 0; r < 8; ++r) inv[r] = (l[r] > 0.f) ? 1.f / l[r] : 0.f;
    #pragma unroll
    for (int t = 0; t < 4; ++t)
        #pragma unroll
        for (int r = 0; r < 8; ++r) {
            const int srow = q0w + r + 8 * hs;
            const int col  = hh * 64 + t * 16 + lm;
            ctx[(size_t)(bb * SEQ + srow) * D_MODEL + col] = f2bf(o[t][r] * inv[r]);
        }
}

// ---------------------------------------------------------------------------
extern "C" void kernel_launch(void* const* d_in, const int* in_sizes, int n_in,
                              void* d_out, int out_size, void* d_ws, size_t ws_size,
                              hipStream_t stream) {
    (void)in_sizes; (void)n_in; (void)out_size; (void)ws_size;
    const float* x  = (const float*)d_in[0];
    const float* Wq = (const float*)d_in[1];
    const float* Wk = (const float*)d_in[2];
    const float* Wv = (const float*)d_in[3];
    const float* Wo = (const float*)d_in[4];
    float* out = (float*)d_out;

    const size_t nelem = (size_t)BATCH * NHEADS * SEQ * DK;   // == B*S*D
    unsigned short* qb = (unsigned short*)d_ws;
    unsigned short* kb = qb + nelem;
    unsigned short* vb = kb + nelem;
    unsigned short* cb = vb + nelem;   // ctx, bf16 [B,S,D]

    const int gemm_blocks = (BATCH * SEQ / 64) * (D_MODEL / 128);  // 1024
    const int attn_blocks = BATCH * NHEADS * (SEQ / 64);           // 2048

    // Q scaled by 1/sqrt(dk) * log2(e): softmax runs in exp2 domain
    const float qscale = 0.125f * 1.44269504088896340736f;
    gemm_kernel<0><<<gemm_blocks, 256, 0, stream>>>(x, Wq, qb, qscale);
    gemm_kernel<0><<<gemm_blocks, 256, 0, stream>>>(x, Wk, kb, 1.0f);
    gemm_kernel<0><<<gemm_blocks, 256, 0, stream>>>(x, Wv, vb, 1.0f);
    attn_kernel<<<attn_blocks, 128, 0, stream>>>(qb, kb, vb, cb);
    gemm_kernel<1><<<gemm_blocks, 256, 0, stream>>>(cb, Wo, out, 1.0f);
}
